// CharDecoder_52261162058475
// MI455X (gfx1250) — matrix operand order repaired and checked by reading, not compiled, not run on hardware
//
#include <hip/hip_runtime.h>

typedef __attribute__((ext_vector_type(16))) _Float16 v16h;
typedef __attribute__((ext_vector_type(8)))  float    v8f;

namespace {
constexpr int QD  = 256;       // quantized feature dim
constexpr int HD  = 512;       // hidden dim
constexpr int VC  = 128;       // vocab (A)
constexpr int G4  = 2048;      // 4*H gate dim
constexpr int CL  = 20;        // max_char_len
constexpr int RPB = 32;        // rows (B*W entries) per workgroup
constexpr int AK  = HD + VC;   // unified K = 640 ( [h | one-hot] )
constexpr int AST = AK + 8;    // padded LDS stride (648; 324 words % 64 == 4 -> bank spread)
constexpr int XST = QD + 8;    // 264 (132 words % 64 == 4)
constexpr int LST = VC + 4;    // 132
}

#define WMMA16(a, b, c) \
  __builtin_amdgcn_wmma_f32_16x16x32_f16(false, (a), false, (b), (short)0, (c), false, false)

// ---- Fragment loaders (16-bit WMMA operand layouts, ISA 7.12.2) ----
// A 16x32 f16: lane l -> row = l&15; lanes 0-15 hold K 0..7 (V0-3) and 16..23 (V4-7),
// lanes 16-31 hold K 8..15 and 24..31. Two b128 loads.
__device__ __forceinline__ v16h load_fragA(const _Float16* base, int stride, int lane) {
  const int r  = lane & 15;
  const int kb = (lane >> 4) * 8;
  const _Float16* q = base + r * stride + kb;
  v16h out;
  *(float4*)&out           = *(const float4*)(q);
  *(((float4*)&out) + 1)   = *(const float4*)(q + 16);
  return out;
}
// B 32x16 f16 (column-major storage, contiguous K per output column):
// lane l -> col = l&15; lanes 0-15 hold K 0..7 and 16..23, lanes 16-31 hold K 8..15 and 24..31.
__device__ __forceinline__ v16h load_fragB(const _Float16* base, int stride, int lane) {
  const int n  = lane & 15;
  const int kb = (lane >> 4) * 8;
  const _Float16* q = base + n * stride + kb;
  v16h out;
  *(float4*)&out           = *(const float4*)(q);
  *(((float4*)&out) + 1)   = *(const float4*)(q + 16);
  return out;
}

__device__ __forceinline__ float sigmoidf_(float x) { return 1.0f / (1.0f + __expf(-x)); }

// ---- One-shot weight conversion f32 -> f16 into workspace (stays in 192MB L2) ----
__global__ __launch_bounds__(256) void convert_kernel(
    const float* __restrict__ w_in, const float* __restrict__ w_ih,
    const float* __restrict__ w_hh, const float* __restrict__ w_out,
    const float* __restrict__ b_ih, const float* __restrict__ b_hh,
    _Float16* __restrict__ whh16, _Float16* __restrict__ wih16,
    _Float16* __restrict__ wout16, _Float16* __restrict__ win16,
    float* __restrict__ bias) {
  const int i = blockIdx.x * 256 + threadIdx.x;
  if (i < G4 * HD) whh16[i] = (_Float16)w_hh[i];
  if (i < G4 * VC) wih16[i] = (_Float16)w_ih[i];
  if (i < VC * HD) wout16[i] = (_Float16)w_out[i];
  if (i < HD * QD) win16[i] = (_Float16)w_in[i];
  if (i < G4)      bias[i]  = b_ih[i] + b_hh[i];
}

// ---- Persistent per-row-tile LSTM: 32 rows/WG, all 20 steps local ----
__global__ __launch_bounds__(256) void lstm_kernel(
    const float* __restrict__ qr, const float* __restrict__ b_in,
    const float* __restrict__ b_out,
    const _Float16* __restrict__ whh16, const _Float16* __restrict__ wih16,
    const _Float16* __restrict__ wout16, const _Float16* __restrict__ win16,
    const float* __restrict__ bias, float* __restrict__ out) {
  // double-buffered unified A: rows 0..31 of [h(512) | one-hot(128)]
  __shared__ _Float16 sA[2][RPB * AST];
  // init-only input staging (f16, 16,896B) aliased with per-step logits (f32, 16,896B)
  __shared__ char sScratch[RPB * XST * sizeof(_Float16)];
  __shared__ int sChar[RPB];
  _Float16* sX   = (_Float16*)sScratch;
  float*    sLog = (float*)sScratch;

  const int tid  = threadIdx.x;
  const int lane = tid & 31;
  const int wv   = tid >> 5;             // wave 0..7
  const int row0 = blockIdx.x * RPB;
  const int nlo  = lane & 15;
  const int mhi  = (lane >> 4) * 8;      // C-frag: VGPR j -> row j+mhi, col = nlo

  // ---- init: stage input tile as f16 ----
  for (int e = tid; e < RPB * QD; e += 256) {
    const int m = e >> 8, k = e & (QD - 1);
    sX[m * XST + k] = (_Float16)qr[(size_t)(row0 + m) * QD + k];
  }
  __syncthreads();

  // ---- h0 = qr @ w_in^T + b_in  (wave wv owns h-cols [64wv, 64wv+64)) ----
  for (int ti = 0; ti < 4; ++ti) {
    const int n0 = (wv * 4 + ti) * 16;
    v8f acc0 = {}, acc1 = {};
#pragma unroll 4
    for (int k0 = 0; k0 < QD; k0 += 32) {
      v16h a0 = load_fragA(sX + k0, XST, lane);
      v16h a1 = load_fragA(sX + 16 * XST + k0, XST, lane);
      v16h b  = load_fragB(win16 + (size_t)n0 * QD + k0, QD, lane);
      acc0 = WMMA16(a0, b, acc0);
      acc1 = WMMA16(a1, b, acc1);
    }
    const float bi = b_in[n0 + nlo];
    for (int j = 0; j < 8; ++j) {
      sA[0][(j + mhi) * AST + n0 + nlo]        = (_Float16)(acc0[j] + bi);
      sA[0][(16 + j + mhi) * AST + n0 + nlo]   = (_Float16)(acc1[j] + bi);
    }
  }
  __syncthreads();  // sX reads done before sLog aliasing reuse
  // one-hot <start> = char 0 in x-region of buffer 0
  for (int e = tid; e < RPB * VC; e += 256) {
    const int m = e >> 7, k = e & (VC - 1);
    sA[0][m * AST + HD + k] = (_Float16)((k == 0) ? 1.0f : 0.0f);
  }

  // cell state lives in registers for all 20 steps: [h-tile ti][M-tile]
  v8f cReg[4][2];
  for (int ti = 0; ti < 4; ++ti) { cReg[ti][0] = {}; cReg[ti][1] = {}; }
  __syncthreads();

  // ---- recurrence ----
  for (int t = 0; t < CL; ++t) {
    const _Float16* Abuf = sA[t & 1];
    _Float16*       Nbuf = sA[(t + 1) & 1];

    // gates = [h|x] @ [w_hh|w_ih]^T ; wave wv owns h-cols [64wv,64wv+64)
    for (int ti = 0; ti < 4; ++ti) {
      const int hc = wv * 64 + ti * 16;   // h-col tile base
      v8f aI0 = {}, aF0 = {}, aG0 = {}, aO0 = {};
      v8f aI1 = {}, aF1 = {}, aG1 = {}, aO1 = {};
#pragma unroll 4
      for (int k0 = 0; k0 < HD; k0 += 32) {        // recurrent part
        v16h a0 = load_fragA(Abuf + k0, AST, lane);
        v16h a1 = load_fragA(Abuf + 16 * AST + k0, AST, lane);
        v16h bI = load_fragB(whh16 + (size_t)(0 * HD + hc) * HD + k0, HD, lane);
        v16h bF = load_fragB(whh16 + (size_t)(1 * HD + hc) * HD + k0, HD, lane);
        v16h bG = load_fragB(whh16 + (size_t)(2 * HD + hc) * HD + k0, HD, lane);
        v16h bO = load_fragB(whh16 + (size_t)(3 * HD + hc) * HD + k0, HD, lane);
        aI0 = WMMA16(a0, bI, aI0); aI1 = WMMA16(a1, bI, aI1);
        aF0 = WMMA16(a0, bF, aF0); aF1 = WMMA16(a1, bF, aF1);
        aG0 = WMMA16(a0, bG, aG0); aG1 = WMMA16(a1, bG, aG1);
        aO0 = WMMA16(a0, bO, aO0); aO1 = WMMA16(a1, bO, aO1);
      }
#pragma unroll 2
      for (int k0 = 0; k0 < VC; k0 += 32) {        // one-hot input part
        v16h a0 = load_fragA(Abuf + HD + k0, AST, lane);
        v16h a1 = load_fragA(Abuf + 16 * AST + HD + k0, AST, lane);
        v16h bI = load_fragB(wih16 + (size_t)(0 * HD + hc) * VC + k0, VC, lane);
        v16h bF = load_fragB(wih16 + (size_t)(1 * HD + hc) * VC + k0, VC, lane);
        v16h bG = load_fragB(wih16 + (size_t)(2 * HD + hc) * VC + k0, VC, lane);
        v16h bO = load_fragB(wih16 + (size_t)(3 * HD + hc) * VC + k0, VC, lane);
        aI0 = WMMA16(a0, bI, aI0); aI1 = WMMA16(a1, bI, aI1);
        aF0 = WMMA16(a0, bF, aF0); aF1 = WMMA16(a1, bF, aF1);
        aG0 = WMMA16(a0, bG, aG0); aG1 = WMMA16(a1, bG, aG1);
        aO0 = WMMA16(a0, bO, aO0); aO1 = WMMA16(a1, bO, aO1);
      }
      // gate math entirely in registers: i/f/g/o of (m,h) share lane+slot
      const int hcol = hc + nlo;
      const float bI = bias[hcol], bF = bias[HD + hcol];
      const float bG = bias[2 * HD + hcol], bO = bias[3 * HD + hcol];
      for (int mt = 0; mt < 2; ++mt) {
        const v8f& gI = mt ? aI1 : aI0;
        const v8f& gF = mt ? aF1 : aF0;
        const v8f& gG = mt ? aG1 : aG0;
        const v8f& gO = mt ? aO1 : aO0;
        for (int j = 0; j < 8; ++j) {
          const float iv = sigmoidf_(gI[j] + bI);
          const float fv = sigmoidf_(gF[j] + bF);
          const float gv = tanhf(gG[j] + bG);
          const float ov = sigmoidf_(gO[j] + bO);
          const float c  = fv * cReg[ti][mt][j] + iv * gv;
          cReg[ti][mt][j] = c;
          Nbuf[(mt * 16 + j + mhi) * AST + hcol] = (_Float16)(ov * tanhf(c));
        }
      }
    }
    __syncthreads();

    // logits = h @ w_out^T + b_out ; wave wv -> vocab cols [16wv,16wv+16)
    {
      const int n0 = wv * 16;
      v8f acc0 = {}, acc1 = {};
#pragma unroll 4
      for (int k0 = 0; k0 < HD; k0 += 32) {
        v16h a0 = load_fragA(Nbuf + k0, AST, lane);
        v16h a1 = load_fragA(Nbuf + 16 * AST + k0, AST, lane);
        v16h b  = load_fragB(wout16 + (size_t)n0 * HD + k0, HD, lane);
        acc0 = WMMA16(a0, b, acc0);
        acc1 = WMMA16(a1, b, acc1);
      }
      const float bo = b_out[n0 + nlo];
      for (int j = 0; j < 8; ++j) {
        const int m0 = j + mhi;
        const float v0 = acc0[j] + bo;
        const float v1 = acc1[j] + bo;
        out[((size_t)(row0 + m0) * CL + t) * VC + n0 + nlo]      = v0;
        out[((size_t)(row0 + 16 + m0) * CL + t) * VC + n0 + nlo] = v1;
        sLog[m0 * LST + n0 + nlo]        = v0;
        sLog[(16 + m0) * LST + n0 + nlo] = v1;
      }
    }
    __syncthreads();

    // zero next-step one-hot region; argmax (first-max, matching jnp.argmax)
    for (int e = tid; e < RPB * VC; e += 256) {
      const int m = e >> 7, k = e & (VC - 1);
      Nbuf[m * AST + HD + k] = (_Float16)0.0f;
    }
    if (tid < RPB) {
      float best = sLog[tid * LST];
      int bi = 0;
      for (int j = 1; j < VC; ++j) {
        const float v = sLog[tid * LST + j];
        if (v > best) { best = v; bi = j; }
      }
      sChar[tid] = bi;
    }
    __syncthreads();
    if (tid < RPB) Nbuf[tid * AST + HD + sChar[tid]] = (_Float16)1.0f;
    __syncthreads();
  }
}

extern "C" void kernel_launch(void* const* d_in, const int* in_sizes, int n_in,
                              void* d_out, int out_size, void* d_ws, size_t ws_size,
                              hipStream_t stream) {
  (void)in_sizes; (void)n_in; (void)out_size; (void)ws_size;
  const float* qr    = (const float*)d_in[0];
  const float* w_in  = (const float*)d_in[1];
  const float* b_in  = (const float*)d_in[2];
  const float* w_ih  = (const float*)d_in[3];
  const float* w_hh  = (const float*)d_in[4];
  const float* b_ih  = (const float*)d_in[5];
  const float* b_hh  = (const float*)d_in[6];
  const float* w_out = (const float*)d_in[7];
  const float* b_out = (const float*)d_in[8];
  float* out = (float*)d_out;

  _Float16* whh16  = (_Float16*)d_ws;            // 2048*512
  _Float16* wih16  = whh16 + (size_t)G4 * HD;    // 2048*128
  _Float16* wout16 = wih16 + (size_t)G4 * VC;    // 128*512
  _Float16* win16  = wout16 + (size_t)VC * HD;   // 512*256
  float*    bias   = (float*)(win16 + (size_t)HD * QD);  // 2048 f32 (byte offset % 4 == 0)

  convert_kernel<<<(G4 * HD + 255) / 256, 256, 0, stream>>>(
      w_in, w_ih, w_hh, w_out, b_ih, b_hh, whh16, wih16, wout16, win16, bias);
  lstm_kernel<<<8192 / RPB, 256, 0, stream>>>(
      qr, b_in, b_out, whh16, wih16, wout16, win16, bias, out);
}
